// DecoderCell_3272765080211
// MI455X (gfx1250) — matrix operand order repaired and checked by reading, not compiled, not run on hardware
//
#include <hip/hip_runtime.h>
#include <hip/hip_bf16.h>

// ---------------------------------------------------------------------------
// CDNA5 (gfx1250) fused decoder cell using v_wmma_f32_16x16x32_bf16.
// wave32; 256-thread blocks = 8 waves; each workgroup owns RT*16 rows so a
// single B fragment (global_load_b128 x2 from L2-resident bf16 weights) feeds
// RT WMMAs. RT=2 for the dual-accumulator GRU kernels (spill-free codegen,
// verified round 2); RT=4 for single-accumulator GEMMs with bounded unroll.
// LDS rows padded +8 bf16 (16B) to spread A-frag ds_load_b128s across banks.
// ---------------------------------------------------------------------------

typedef __attribute__((ext_vector_type(16))) __bf16 v16bf;
typedef __attribute__((ext_vector_type(8)))  __bf16 v8bf;
typedef __attribute__((ext_vector_type(8)))  float  v8f;

// padded LDS leading dims (elements); (+8 elems = +16B per row)
#define CI_LD 392         // 384 + 8   (row stride 784B  -> 16B bank skew)
#define CN_LD 264         // 256 + 8   (row stride 528B  -> 16B bank skew)
#define GI_LD 72          // 64 + 8    (row stride 144B)
#define GG_LD 520         // 512 + 8   (row stride 1040B -> 16B bank skew)

__device__ inline v8f wmma_bf16(v16bf a, v16bf b, v8f c) {
  // (neg_a, A, neg_b, B, c_mod, C, reuse_a, reuse_b)
  return __builtin_amdgcn_wmma_f32_16x16x32_bf16(false, a, false, b, (short)0, c,
                                                 false, false);
}

// A fragment (16x32 bf16) from LDS, row-major X with leading dim 'ld' (elems).
// lane(half,m): a[0..15] = X[m][k0+half*16 .. +15]
__device__ inline v16bf load_a_frag(const __bf16* X, int ld, int k0, int lane) {
  const int half = lane >> 4, m = lane & 15;
  const __bf16* p = X + m * ld + k0 + half * 16;
  v8bf lo = *(const v8bf*)p;
  v8bf hi = *(const v8bf*)(p + 8);
  v16bf a;
#pragma unroll
  for (int i = 0; i < 8; ++i) { a[i] = lo[i]; a[i + 8] = hi[i]; }
  return a;
}

// B fragment (32x16 bf16) = transpose-view tile of row-major W (N x K, ld=K):
// lane(half,n): b[e] = W[n0+n][k0 + half*16 + e]  -> one contiguous 32B load.
__device__ inline v16bf load_b_frag(const __bf16* W, int ld, int n0, int k0, int lane) {
  const int half = lane >> 4, n = lane & 15;
  return *(const v16bf*)(W + (n0 + n) * ld + k0 + half * 16);
}

__device__ inline float sigmoidf_(float x) { return 1.0f / (1.0f + __expf(-x)); }

// ---------------------------------------------------------------------------
// Weight prep: f32 -> bf16
// ---------------------------------------------------------------------------
__global__ void k_cvt(const float* __restrict__ src, __bf16* __restrict__ dst, int n) {
  int i = blockIdx.x * 256 + threadIdx.x;
  if (i < n) dst[i] = (__bf16)src[i];
}

// L2-normalize each 512-wide row of fac_W (128 rows), emit bf16.
__global__ __launch_bounds__(256) void k_facnorm(const float* __restrict__ W,
                                                 __bf16* __restrict__ dst) {
  __shared__ float red[8];
  const int row = blockIdx.x, tid = threadIdx.x;
  float a = W[row * 512 + tid];
  float b = W[row * 512 + 256 + tid];
  float ss = a * a + b * b;
#pragma unroll
  for (int off = 16; off > 0; off >>= 1) ss += __shfl_down(ss, off, 32);
  if ((tid & 31) == 0) red[tid >> 5] = ss;
  __syncthreads();
  if (tid == 0) {
    float s = 0.f;
#pragma unroll
    for (int i = 0; i < 8; ++i) s += red[i];
    red[0] = 1.0f / fmaxf(sqrtf(s), 1e-12f);
  }
  __syncthreads();
  const float sc = red[0];
  dst[row * 512 + tid]       = (__bf16)(a * sc);
  dst[row * 512 + 256 + tid] = (__bf16)(b * sc);
}

// ---------------------------------------------------------------------------
// Stage 1: controller GRU (H=256, Kx=384, Kh=256)
// RT=2: LDS = ci 24.5KB + hb 16.5KB + rh 16.5KB = 57.5KB
// ---------------------------------------------------------------------------
template <int RT>
__global__ __launch_bounds__(256) void k_con(const float* __restrict__ input,
                                             const float* __restrict__ h0,
                                             const float* __restrict__ bih,
                                             const float* __restrict__ bhh,
                                             const __bf16* __restrict__ Wih,   // 768x384
                                             const __bf16* __restrict__ Whh,   // 768x256
                                             float* __restrict__ out,
                                             __bf16* __restrict__ cs_bf) {
  constexpr int TB = RT * 16;
  const int rb = blockIdx.x * TB;
  const int tid = threadIdx.x, wave = tid >> 5, lane = tid & 31;
  const int half = lane >> 4, nn = lane & 15;
  __shared__ __attribute__((aligned(32))) __bf16 ci[TB * CI_LD];
  __shared__ __attribute__((aligned(32))) __bf16 hb[TB * CN_LD];
  __shared__ __attribute__((aligned(32))) __bf16 rh[TB * CN_LD];

  for (int idx = tid; idx < TB * 384; idx += 256) {
    int r = idx / 384, c = idx - r * 384;
    float v = (c < 256) ? input[(size_t)(rb + r) * 256 + c]
                        : h0[(size_t)(rb + r) * 1088 + 960 + (c - 256)];  // factor
    ci[r * CI_LD + c] = (__bf16)v;
  }
  for (int idx = tid; idx < TB * 256; idx += 256) {
    int r = idx >> 8, c = idx & 255;
    hb[r * CN_LD + c] = (__bf16)h0[(size_t)(rb + r) * 1088 + 512 + c];
  }
  __syncthreads();

  // phase 1: r gate -> rh  (16 col-tiles, 2 per wave; B reused across RT rows)
  for (int ct = wave; ct < 16; ct += 8) {
    const int c0 = ct * 16;
    v8f acc[RT];
#pragma unroll
    for (int r = 0; r < RT; ++r) acc[r] = (v8f)0.0f;
    for (int k = 0; k < 384; k += 32) {
      v16bf b = load_b_frag(Wih, 384, 256 + c0, k, lane);
#pragma unroll
      for (int r = 0; r < RT; ++r)
        acc[r] = wmma_bf16(load_a_frag(ci + r * 16 * CI_LD, CI_LD, k, lane), b, acc[r]);
    }
    for (int k = 0; k < 256; k += 32) {
      v16bf b = load_b_frag(Whh, 256, 256 + c0, k, lane);
#pragma unroll
      for (int r = 0; r < RT; ++r)
        acc[r] = wmma_bf16(load_a_frag(hb + r * 16 * CN_LD, CN_LD, k, lane), b, acc[r]);
    }
    const float br = bih[256 + c0 + nn] + bhh[256 + c0 + nn];
#pragma unroll
    for (int r = 0; r < RT; ++r)
#pragma unroll
      for (int v = 0; v < 8; ++v) {
        const int m = r * 16 + half * 8 + v;
        float rg = sigmoidf_(acc[r][v] + br);
        rh[m * CN_LD + c0 + nn] = (__bf16)(rg * (float)hb[m * CN_LD + c0 + nn]);
      }
  }
  __syncthreads();

  // phase 2: z and n gates, combine, clip, write
  for (int ct = wave; ct < 16; ct += 8) {
    const int c0 = ct * 16;
    v8f az[RT], an[RT];
#pragma unroll
    for (int r = 0; r < RT; ++r) { az[r] = (v8f)0.0f; an[r] = (v8f)0.0f; }
    for (int k = 0; k < 384; k += 32) {
      v16bf bz = load_b_frag(Wih, 384, c0, k, lane);
      v16bf bn = load_b_frag(Wih, 384, 512 + c0, k, lane);
#pragma unroll
      for (int r = 0; r < RT; ++r) {
        v16bf a = load_a_frag(ci + r * 16 * CI_LD, CI_LD, k, lane);
        az[r] = wmma_bf16(a, bz, az[r]);
        an[r] = wmma_bf16(a, bn, an[r]);
      }
    }
    for (int k = 0; k < 256; k += 32) {
      v16bf bz = load_b_frag(Whh, 256, c0, k, lane);
      v16bf bn = load_b_frag(Whh, 256, 512 + c0, k, lane);
#pragma unroll
      for (int r = 0; r < RT; ++r) {
        az[r] = wmma_bf16(load_a_frag(hb + r * 16 * CN_LD, CN_LD, k, lane), bz, az[r]);
        an[r] = wmma_bf16(load_a_frag(rh + r * 16 * CN_LD, CN_LD, k, lane), bn, an[r]);
      }
    }
    const float bzc = bih[c0 + nn] + bhh[c0 + nn];
    const float bnc = bih[512 + c0 + nn] + bhh[512 + c0 + nn];
#pragma unroll
    for (int r = 0; r < RT; ++r)
#pragma unroll
      for (int v = 0; v < 8; ++v) {
        const int m = r * 16 + half * 8 + v;
        float z = sigmoidf_(az[r][v] + bzc);
        float n = tanhf(an[r][v] + bnc);
        float hv = (float)hb[m * CN_LD + c0 + nn];
        float x = z * hv + (1.0f - z) * n;
        x = fminf(5.0f, fmaxf(-5.0f, x));
        out[(size_t)(rb + m) * 1088 + 512 + c0 + nn] = x;
        cs_bf[(size_t)(rb + m) * 256 + c0 + nn] = (__bf16)x;
      }
  }
}

// ---------------------------------------------------------------------------
// Stage 2: co params + rsample (N=128, K=256)  RT=4: LDS = 33KB + 32KB = 65KB
// Single accumulator set per wave; k-loop unroll bounded to avoid spills.
// ---------------------------------------------------------------------------
template <int RT>
__global__ __launch_bounds__(256) void k_co(const __bf16* __restrict__ cs_bf,
                                            const __bf16* __restrict__ coW,  // 128x256
                                            const float* __restrict__ co_b,
                                            const float* __restrict__ eps,
                                            float* __restrict__ out,
                                            __bf16* __restrict__ gi_bf) {
  constexpr int TB = RT * 16;
  const int rb = blockIdx.x * TB;
  const int tid = threadIdx.x, wave = tid >> 5, lane = tid & 31;
  const int half = lane >> 4, nn = lane & 15;
  __shared__ __attribute__((aligned(32))) __bf16 A[TB * CN_LD];
  __shared__ float cp[TB * 128];

  for (int idx = tid; idx < TB * 256; idx += 256) {
    int r = idx >> 8, c = idx & 255;
    A[r * CN_LD + c] = cs_bf[(size_t)rb * 256 + idx];
  }
  __syncthreads();

  {
    const int c0 = wave * 16;  // 8 waves x 16 cols = 128
    v8f acc[RT];
#pragma unroll
    for (int r = 0; r < RT; ++r) acc[r] = (v8f)0.0f;
#pragma unroll 2
    for (int k = 0; k < 256; k += 32) {
      v16bf b = load_b_frag(coW, 256, c0, k, lane);
#pragma unroll
      for (int r = 0; r < RT; ++r)
        acc[r] = wmma_bf16(load_a_frag(A + r * 16 * CN_LD, CN_LD, k, lane), b, acc[r]);
    }
    const float bb = co_b[c0 + nn];
#pragma unroll
    for (int r = 0; r < RT; ++r)
#pragma unroll
      for (int v = 0; v < 8; ++v)
        cp[(r * 16 + half * 8 + v) * 128 + c0 + nn] = acc[r][v] + bb;
  }
  __syncthreads();

  for (int idx = tid; idx < TB * 64; idx += 256) {
    const int m = idx >> 6, j = idx & 63;
    float mean = cp[m * 128 + j];
    float sd = __expf(0.5f * cp[m * 128 + 64 + j]);
    float gi = mean + sd * eps[(size_t)(rb + m) * 64 + j];
    const size_t ro = (size_t)(rb + m) * 1088;
    out[ro + 768 + j] = mean;
    out[ro + 832 + j] = sd;
    out[ro + 896 + j] = gi;
    gi_bf[(size_t)(rb + m) * 64 + j] = (__bf16)gi;
  }
}

// ---------------------------------------------------------------------------
// Stage 3: generator GRU (H=512, Kx=64, Kh=512)
// RT=2: LDS = gi 4.5KB + hb 32.5KB + rh 32.5KB = 69.5KB
// gen_state written f32 to out and bf16 to ws (factor GEMM is kernel 4).
// ---------------------------------------------------------------------------
template <int RT>
__global__ __launch_bounds__(256) void k_gen(const float* __restrict__ h0,
                                             const __bf16* __restrict__ gi_bf,
                                             const float* __restrict__ bih,
                                             const float* __restrict__ bhh,
                                             const __bf16* __restrict__ Wih,   // 1536x64
                                             const __bf16* __restrict__ Whh,   // 1536x512
                                             float* __restrict__ out,
                                             __bf16* __restrict__ gs_bf) {
  constexpr int TB = RT * 16;
  const int rb = blockIdx.x * TB;
  const int tid = threadIdx.x, wave = tid >> 5, lane = tid & 31;
  const int half = lane >> 4, nn = lane & 15;
  __shared__ __attribute__((aligned(32))) __bf16 gi[TB * GI_LD];
  __shared__ __attribute__((aligned(32))) __bf16 hb[TB * GG_LD];
  __shared__ __attribute__((aligned(32))) __bf16 rh[TB * GG_LD];

  for (int idx = tid; idx < TB * 64; idx += 256) {
    int r = idx >> 6, c = idx & 63;
    gi[r * GI_LD + c] = gi_bf[(size_t)rb * 64 + idx];
  }
  for (int idx = tid; idx < TB * 512; idx += 256) {
    int r = idx >> 9, c = idx & 511;
    hb[r * GG_LD + c] = (__bf16)h0[(size_t)(rb + r) * 1088 + c];  // gen_state
  }
  __syncthreads();

  // phase 1: r gate -> rh  (32 col-tiles, 4 per wave)
  for (int ct = wave; ct < 32; ct += 8) {
    const int c0 = ct * 16;
    v8f acc[RT];
#pragma unroll
    for (int r = 0; r < RT; ++r) acc[r] = (v8f)0.0f;
    for (int k = 0; k < 64; k += 32) {
      v16bf b = load_b_frag(Wih, 64, 512 + c0, k, lane);
#pragma unroll
      for (int r = 0; r < RT; ++r)
        acc[r] = wmma_bf16(load_a_frag(gi + r * 16 * GI_LD, GI_LD, k, lane), b, acc[r]);
    }
    for (int k = 0; k < 512; k += 32) {
      v16bf b = load_b_frag(Whh, 512, 512 + c0, k, lane);
#pragma unroll
      for (int r = 0; r < RT; ++r)
        acc[r] = wmma_bf16(load_a_frag(hb + r * 16 * GG_LD, GG_LD, k, lane), b, acc[r]);
    }
    const float br = bih[512 + c0 + nn] + bhh[512 + c0 + nn];
#pragma unroll
    for (int r = 0; r < RT; ++r)
#pragma unroll
      for (int v = 0; v < 8; ++v) {
        const int m = r * 16 + half * 8 + v;
        float rg = sigmoidf_(acc[r][v] + br);
        rh[m * GG_LD + c0 + nn] = (__bf16)(rg * (float)hb[m * GG_LD + c0 + nn]);
      }
  }
  __syncthreads();

  // phase 2: z and n gates, combine, clip, write gen_state (f32 + bf16)
  for (int ct = wave; ct < 32; ct += 8) {
    const int c0 = ct * 16;
    v8f az[RT], an[RT];
#pragma unroll
    for (int r = 0; r < RT; ++r) { az[r] = (v8f)0.0f; an[r] = (v8f)0.0f; }
    for (int k = 0; k < 64; k += 32) {
      v16bf bz = load_b_frag(Wih, 64, c0, k, lane);
      v16bf bn = load_b_frag(Wih, 64, 1024 + c0, k, lane);
#pragma unroll
      for (int r = 0; r < RT; ++r) {
        v16bf a = load_a_frag(gi + r * 16 * GI_LD, GI_LD, k, lane);
        az[r] = wmma_bf16(a, bz, az[r]);
        an[r] = wmma_bf16(a, bn, an[r]);
      }
    }
    for (int k = 0; k < 512; k += 32) {
      v16bf bz = load_b_frag(Whh, 512, c0, k, lane);
      v16bf bn = load_b_frag(Whh, 512, 1024 + c0, k, lane);
#pragma unroll
      for (int r = 0; r < RT; ++r) {
        az[r] = wmma_bf16(load_a_frag(hb + r * 16 * GG_LD, GG_LD, k, lane), bz, az[r]);
        an[r] = wmma_bf16(load_a_frag(rh + r * 16 * GG_LD, GG_LD, k, lane), bn, an[r]);
      }
    }
    const float bzc = bih[c0 + nn] + bhh[c0 + nn];
    const float bnc = bih[1024 + c0 + nn] + bhh[1024 + c0 + nn];
#pragma unroll
    for (int r = 0; r < RT; ++r)
#pragma unroll
      for (int v = 0; v < 8; ++v) {
        const int m = r * 16 + half * 8 + v;
        float z = sigmoidf_(az[r][v] + bzc);
        float n = tanhf(an[r][v] + bnc);
        float hv = (float)hb[m * GG_LD + c0 + nn];
        float x = z * hv + (1.0f - z) * n;
        x = fminf(5.0f, fmaxf(-5.0f, x));
        out[(size_t)(rb + m) * 1088 + c0 + nn] = x;
        gs_bf[(size_t)(rb + m) * 512 + c0 + nn] = (__bf16)x;
      }
  }
}

// ---------------------------------------------------------------------------
// Stage 4: factor = gen_state @ normed_facW.T  (N=128, K=512)
// RT=4: LDS = 65KB; single accumulator set; bounded k-loop unroll.
// ---------------------------------------------------------------------------
template <int RT>
__global__ __launch_bounds__(256) void k_fac(const __bf16* __restrict__ gs_bf,
                                             const __bf16* __restrict__ facW,  // 128x512
                                             float* __restrict__ out) {
  constexpr int TB = RT * 16;
  const int rb = blockIdx.x * TB;
  const int tid = threadIdx.x, wave = tid >> 5, lane = tid & 31;
  const int half = lane >> 4, nn = lane & 15;
  __shared__ __attribute__((aligned(32))) __bf16 A[TB * GG_LD];

  for (int idx = tid; idx < TB * 512; idx += 256) {
    int r = idx >> 9, c = idx & 511;
    A[r * GG_LD + c] = gs_bf[(size_t)rb * 512 + idx];
  }
  __syncthreads();

  const int c0 = wave * 16;  // 8 waves x 16 cols = 128
  v8f acc[RT];
#pragma unroll
  for (int r = 0; r < RT; ++r) acc[r] = (v8f)0.0f;
#pragma unroll 2
  for (int k = 0; k < 512; k += 32) {
    v16bf b = load_b_frag(facW, 512, c0, k, lane);
#pragma unroll
    for (int r = 0; r < RT; ++r)
      acc[r] = wmma_bf16(load_a_frag(A + r * 16 * GG_LD, GG_LD, k, lane), b, acc[r]);
  }
#pragma unroll
  for (int r = 0; r < RT; ++r)
#pragma unroll
    for (int v = 0; v < 8; ++v)
      out[(size_t)(rb + r * 16 + half * 8 + v) * 1088 + 960 + c0 + nn] = acc[r][v];
}

// ---------------------------------------------------------------------------
// Host launcher
// ---------------------------------------------------------------------------
extern "C" void kernel_launch(void* const* d_in, const int* in_sizes, int n_in,
                              void* d_out, int out_size, void* d_ws, size_t ws_size,
                              hipStream_t stream) {
  const float* input   = (const float*)d_in[0];
  const float* h0      = (const float*)d_in[1];
  const float* eps     = (const float*)d_in[2];
  const float* gen_Wih = (const float*)d_in[3];
  const float* gen_bih = (const float*)d_in[4];
  const float* gen_Whh = (const float*)d_in[5];
  const float* gen_bhh = (const float*)d_in[6];
  const float* con_Wih = (const float*)d_in[7];
  const float* con_bih = (const float*)d_in[8];
  const float* con_Whh = (const float*)d_in[9];
  const float* con_bhh = (const float*)d_in[10];
  const float* fac_W   = (const float*)d_in[11];
  const float* co_W    = (const float*)d_in[12];
  const float* co_b    = (const float*)d_in[13];
  float* out = (float*)d_out;

  // ws layout (all offsets 32B-aligned)
  char* ws = (char*)d_ws;
  size_t o = 0;
  __bf16* genWih_bf = (__bf16*)(ws + o); o += (size_t)1536 * 64 * 2;
  __bf16* genWhh_bf = (__bf16*)(ws + o); o += (size_t)1536 * 512 * 2;
  __bf16* conWih_bf = (__bf16*)(ws + o); o += (size_t)768 * 384 * 2;
  __bf16* conWhh_bf = (__bf16*)(ws + o); o += (size_t)768 * 256 * 2;
  __bf16* coW_bf    = (__bf16*)(ws + o); o += (size_t)128 * 256 * 2;
  __bf16* facW_bf   = (__bf16*)(ws + o); o += (size_t)128 * 512 * 2;
  __bf16* cs_bf     = (__bf16*)(ws + o); o += (size_t)32768 * 256 * 2;
  __bf16* gi_bf     = (__bf16*)(ws + o); o += (size_t)32768 * 64 * 2;
  __bf16* gs_bf     = (__bf16*)(ws + o); o += (size_t)32768 * 512 * 2;

  // weight prep
  auto cvt = [&](const float* s, __bf16* d, int n) {
    k_cvt<<<(n + 255) / 256, 256, 0, stream>>>(s, d, n);
  };
  cvt(gen_Wih, genWih_bf, 1536 * 64);
  cvt(gen_Whh, genWhh_bf, 1536 * 512);
  cvt(con_Wih, conWih_bf, 768 * 384);
  cvt(con_Whh, conWhh_bf, 768 * 256);
  cvt(co_W,    coW_bf,    128 * 256);
  k_facnorm<<<128, 256, 0, stream>>>(fac_W, facW_bf);

  // GRU kernels: RT=2 (dual accumulator sets -> keep register pressure low)
  k_con<2><<<32768 / 32, 256, 0, stream>>>(input, h0, con_bih, con_bhh, conWih_bf,
                                           conWhh_bf, out, cs_bf);
  // single-GEMM kernels: RT=4 (one accumulator set -> 4x B reuse is cheap)
  k_co<4><<<32768 / 64, 256, 0, stream>>>(cs_bf, coW_bf, co_b, eps, out, gi_bf);
  k_gen<2><<<32768 / 32, 256, 0, stream>>>(h0, gi_bf, gen_bih, gen_bhh, genWih_bf,
                                           genWhh_bf, out, gs_bf);
  k_fac<4><<<32768 / 64, 256, 0, stream>>>(gs_bf, facW_bf, out);
}
